// MultiHeadAttention_22239340659162
// MI455X (gfx1250) — matrix operand order repaired and checked by reading, not compiled, not run on hardware
//
#include <hip/hip_runtime.h>
#include <hip/hip_bf16.h>
#include <math.h>
#include <stdint.h>

typedef __attribute__((ext_vector_type(16))) _Float16 v16h;
typedef __attribute__((ext_vector_type(8)))  _Float16 v8h;
typedef __attribute__((ext_vector_type(2)))  _Float16 v2h;
typedef __attribute__((ext_vector_type(8)))  float    v8f;

static constexpr int Bb = 2;
static constexpr int S  = 2048;
static constexpr int D  = 2048;
static constexpr int H  = 16;
static constexpr int DH = 128;

__device__ __forceinline__ v8f wmma_f16(v16h a, v16h b, v8f c) {
  return __builtin_amdgcn_wmma_f32_16x16x32_f16(
      /*neg_a=*/false, a, /*neg_b=*/false, b,
      /*c_mod=*/(short)0, c, /*reuse_a=*/false, /*reuse_b=*/false);
}

// Async global->LDS copy, 16B per lane (GLOBAL_LOAD_ASYNC_TO_LDS_B128).
// Tracked on ASYNCcnt; data bypasses VGPRs entirely.
__device__ __forceinline__ void async_copy_b128(void* lds_ptr, const void* gptr) {
  uint32_t lds_off = (uint32_t)(uintptr_t)lds_ptr;   // low 32 bits = LDS offset
  uint64_t gaddr   = (uint64_t)(uintptr_t)gptr;
  asm volatile("global_load_async_to_lds_b128 %0, %1, off"
               :: "v"(lds_off), "v"(gaddr) : "memory");
}
__device__ __forceinline__ void wait_async0() {
  asm volatile("s_wait_asynccnt 0" ::: "memory");
}

// A fragment: 16x32 f16, rows m0..m0+15, cols k0..k0+31 from row-major LDS tile.
// Layout (ISA 7.12.2): lane row = lane&15; halves [0..7]+half*8, [16..23]+half*8.
__device__ __forceinline__ v16h load_a_frag(const _Float16* tile, int ld,
                                            int m0, int k0, int lane) {
  int row  = lane & 15;
  int half = (lane >> 4) & 1;
  const _Float16* p = tile + (size_t)(m0 + row) * ld + k0 + half * 8;
  v8h lo = *(const v8h*)p;
  v8h hi = *(const v8h*)(p + 16);
  v16h a;
#pragma unroll
  for (int i = 0; i < 8; ++i) { a[i] = lo[i]; a[i + 8] = hi[i]; }
  return a;
}

// B fragment: 32x16 f16 (KxN) from a [K][N] row-major LDS tile.
__device__ __forceinline__ v16h load_b_frag_kn(const _Float16* tile, int ld,
                                               int k0, int n0, int lane) {
  int half = (lane >> 4) & 1;
  const _Float16* p = tile + (size_t)(k0 + half * 8) * ld + n0 + (lane & 15);
  v16h b;
#pragma unroll
  for (int i = 0; i < 8; ++i) { b[i] = p[(size_t)i * ld]; b[i + 8] = p[(size_t)(i + 16) * ld]; }
  return b;
}

__device__ __forceinline__ float half_max(float v) {
#pragma unroll
  for (int m = 1; m <= 8; m <<= 1) v = fmaxf(v, __shfl_xor(v, m, 32));
  return v;
}
__device__ __forceinline__ float half_sum(float v) {
#pragma unroll
  for (int m = 1; m <= 8; m <<= 1) v += __shfl_xor(v, m, 32);
  return v;
}

// ---------------------------------------------------------------------------
// Kernel 1: Y = X @ W^T, output head-split (B,H,S,DH) as f16.
// Block tile 64(M) x 128(N), 8 waves of 32x32, K-step 64 (2 WMMA chains/stage).
// W tile staged TRANSPOSED in LDS -> A-fragment reads are 16B ds_loads, B-fragment reads are strided 2B loads.
// ---------------------------------------------------------------------------
__global__ __launch_bounds__(256) void mha_proj(
    const float* __restrict__ qin, const float* __restrict__ kvin,
    const float* __restrict__ Wq, const float* __restrict__ Wk,
    const float* __restrict__ Wv,
    _Float16* __restrict__ Qh, _Float16* __restrict__ Kh,
    _Float16* __restrict__ Vh) {
  __shared__ _Float16 As[64][72];    // [m][k]  (k-step 64, pad 8)
  __shared__ _Float16 Bs[64][136];   // [k][n]  transposed W tile (pad 8)

  const int z = blockIdx.z;
  const float* X = (z == 0) ? qin : kvin;
  const float* W = (z == 0) ? Wq : ((z == 1) ? Wk : Wv);
  _Float16*    Y = (z == 0) ? Qh : ((z == 1) ? Kh : Vh);

  const int m0 = blockIdx.y * 64;
  const int n0 = blockIdx.x * 128;
  const int tid = threadIdx.x, lane = tid & 31, wid = tid >> 5;
  const int wm = (wid >> 2) * 32;   // 0 or 32
  const int wn = (wid & 3) * 32;    // 0,32,64,96

  v8f acc[2][2] = {};

  for (int k0 = 0; k0 < D; k0 += 64) {
    // Stage A: 64x64 f32 -> f16, row-major (thread: 16 contiguous cols)
    {
      int r = tid >> 2, cb = (tid & 3) * 16;
      const float* src = X + (size_t)(m0 + r) * D + k0 + cb;
      _Float16* dst = &As[r][cb];
#pragma unroll
      for (int q = 0; q < 4; ++q) {
        float4 f = *(const float4*)(src + 4 * q);
        dst[4 * q + 0] = (_Float16)f.x; dst[4 * q + 1] = (_Float16)f.y;
        dst[4 * q + 2] = (_Float16)f.z; dst[4 * q + 3] = (_Float16)f.w;
      }
    }
    // Stage B transposed: thread handles 2 adjacent n-rows x 16 k, packs
    // (W[n,k], W[n+1,k]) pairs -> one ds_store_b32 each.
    {
      int n  = 2 * (tid >> 2);          // 0..126
      int kc = (tid & 3) * 16;          // 0..48
      const float* s0 = W + (size_t)(n0 + n) * D + k0 + kc;
      const float* s1 = s0 + D;
#pragma unroll
      for (int q = 0; q < 4; ++q) {
        float4 f0 = *(const float4*)(s0 + 4 * q);
        float4 f1 = *(const float4*)(s1 + 4 * q);
        v2h p0 = { (_Float16)f0.x, (_Float16)f1.x };
        v2h p1 = { (_Float16)f0.y, (_Float16)f1.y };
        v2h p2 = { (_Float16)f0.z, (_Float16)f1.z };
        v2h p3 = { (_Float16)f0.w, (_Float16)f1.w };
        *(v2h*)&Bs[kc + 4 * q + 0][n] = p0;
        *(v2h*)&Bs[kc + 4 * q + 1][n] = p1;
        *(v2h*)&Bs[kc + 4 * q + 2][n] = p2;
        *(v2h*)&Bs[kc + 4 * q + 3][n] = p3;
      }
    }
    // Prefetch next K-panel while this stage computes (global_prefetch_b8).
    if (k0 + 64 < D) {
      __builtin_prefetch(X + (size_t)(m0 + (tid >> 2)) * D + k0 + 64, 0, 1);
      __builtin_prefetch(W + (size_t)(n0 + (tid >> 1)) * D + k0 + 64, 0, 1);
    }
    __syncthreads();

#pragma unroll
    for (int kk = 0; kk < 2; ++kk) {
      v16h a0 = load_a_frag(&As[0][0], 72, wm + 0,  kk * 32, lane);
      v16h a1 = load_a_frag(&As[0][0], 72, wm + 16, kk * 32, lane);
      v16h b0 = load_b_frag_kn(&Bs[0][0], 136, kk * 32, wn + 0,  lane);
      v16h b1 = load_b_frag_kn(&Bs[0][0], 136, kk * 32, wn + 16, lane);
      acc[0][0] = wmma_f16(a0, b0, acc[0][0]);
      acc[0][1] = wmma_f16(a0, b1, acc[0][1]);
      acc[1][0] = wmma_f16(a1, b0, acc[1][0]);
      acc[1][1] = wmma_f16(a1, b1, acc[1][1]);
    }
    __syncthreads();
  }

  // Store head-split f16: Y[(b*H+h)*S + s][dh]
  const int halfw = (lane >> 4) & 1, ln = lane & 15;
#pragma unroll
  for (int mf = 0; mf < 2; ++mf)
#pragma unroll
    for (int nf = 0; nf < 2; ++nf)
#pragma unroll
      for (int r = 0; r < 8; ++r) {
        int m = m0 + wm + mf * 16 + r + halfw * 8;
        int n = n0 + wn + nf * 16 + ln;
        int b  = m >> 11, s  = m & (S - 1);
        int h  = n >> 7,  dh = n & (DH - 1);
        Y[(((size_t)(b * H + h) * S + s) * DH) + dh] = (_Float16)acc[mf][nf][r];
      }
}

// ---------------------------------------------------------------------------
// Kernel 2: flash attention. One block per (b, h, 64-query tile); 4 waves,
// each wave owns 16 query rows. Q and V tiles staged with async global->LDS
// (ASYNCcnt); K chunk transposed through VGPRs into Kt[d][key].
// ---------------------------------------------------------------------------
__global__ __launch_bounds__(128) void mha_attn(
    const _Float16* __restrict__ Qh, const _Float16* __restrict__ Kh,
    const _Float16* __restrict__ Vh, const unsigned char* __restrict__ maskb,
    float* __restrict__ attn) {
  __shared__ _Float16 Qs[64][136];
  __shared__ _Float16 Kt[128][72];     // transposed K chunk: [d][key]
  __shared__ _Float16 Vs[64][136];     // [key][d]
  __shared__ _Float16 Ps[4][16][72];   // per-wave P scratch

  const int b = blockIdx.z, h = blockIdx.y, q0 = blockIdx.x * 64;
  const int tid = threadIdx.x, lane = tid & 31, wid = tid >> 5;
  const int halfw = (lane >> 4) & 1, ln = lane & 15;

  const size_t head = ((size_t)(b * H + h)) * S * DH;
  const _Float16* Qb = Qh + head;
  const _Float16* Kb = Kh + head;
  const _Float16* Vb = Vh + head;
  const unsigned char* mrow = maskb + (size_t)b * S;

  // Q tile 64x128: async copy straight into LDS (raw f16 bytes).
#pragma unroll
  for (int i = 0; i < 8; ++i) {
    int vi = tid + i * 128;
    int r = vi >> 4, cb = (vi & 15) * 8;
    async_copy_b128(&Qs[r][cb], Qb + (size_t)(q0 + r) * DH + cb);
  }
  wait_async0();
  __syncthreads();

  v16h qa[4];
#pragma unroll
  for (int kf = 0; kf < 4; ++kf)
    qa[kf] = load_a_frag(&Qs[0][0], 136, wid * 16, kf * 32, lane);

  const float scale = 0.08838834764831845f;  // 1/sqrt(128)
  float m_st[8], l_st[8];
#pragma unroll
  for (int r = 0; r < 8; ++r) { m_st[r] = -INFINITY; l_st[r] = 0.f; }
  v8f o[8] = {};

  for (int kb = 0; kb < S; kb += 64) {
    __syncthreads();
    // V chunk: async copy [key][d] (no layout change needed).
#pragma unroll
    for (int i = 0; i < 8; ++i) {
      int vi = tid + i * 128;
      int r = vi >> 4, cb = (vi & 15) * 8;
      async_copy_b128(&Vs[r][cb], Vb + (size_t)(kb + r) * DH + cb);
    }
    // K chunk: transpose into Kt[d][key]; pack adjacent-key pairs -> b32 stores
#pragma unroll
    for (int i = 0; i < 4; ++i) {
      int idx = tid + i * 128;          // 0..511
      int kp  = idx >> 4;               // key pair 0..31
      int dc  = (idx & 15) * 8;         // d chunk
      v8h r0 = *(const v8h*)(Kb + (size_t)(kb + 2 * kp)     * DH + dc);
      v8h r1 = *(const v8h*)(Kb + (size_t)(kb + 2 * kp + 1) * DH + dc);
#pragma unroll
      for (int e = 0; e < 8; ++e) {
        v2h pp = { r0[e], r1[e] };
        *(v2h*)&Kt[dc + e][2 * kp] = pp;
      }
    }
    // Prefetch next K chunk (global_prefetch_b8).
    if (kb + 64 < S) {
      int pr = tid >> 1, pc = (tid & 1) * 64;
      __builtin_prefetch(Kb + (size_t)(kb + 64 + pr) * DH + pc, 0, 1);
    }
    wait_async0();
    __syncthreads();

    // S = Q K^T  (4 N-frags x 4 K-chain), B frags contiguous from Kt
    v8f sc[4];
#pragma unroll
    for (int j = 0; j < 4; ++j) {
      v8f c = {};
#pragma unroll
      for (int kf = 0; kf < 4; ++kf) {
        v16h bk = load_b_frag_kn(&Kt[0][0], 72, kf * 32, j * 16, lane);
        c = wmma_f16(qa[kf], bk, c);
      }
      sc[j] = c;
    }

    // scale + mask
#pragma unroll
    for (int j = 0; j < 4; ++j) {
      bool mm = mrow[kb + j * 16 + ln] != 0;
#pragma unroll
      for (int r = 0; r < 8; ++r) {
        float v = sc[j][r] * scale;
        sc[j][r] = mm ? -INFINITY : v;
      }
    }

    // online softmax (row = r + halfw*8 of this wave's 16-row tile)
    float m_new[8], alpha[8], rs[8];
#pragma unroll
    for (int r = 0; r < 8; ++r) {
      float v = fmaxf(fmaxf(sc[0][r], sc[1][r]), fmaxf(sc[2][r], sc[3][r]));
      v = half_max(v);
      float mn = fmaxf(m_st[r], v);
      m_new[r] = mn;
      alpha[r] = (mn == -INFINITY) ? 1.0f : __expf(m_st[r] - mn);
      rs[r] = 0.f;
    }
#pragma unroll
    for (int j = 0; j < 4; ++j)
#pragma unroll
      for (int r = 0; r < 8; ++r) {
        float sv = sc[j][r];
        float p = (sv == -INFINITY) ? 0.f : __expf(sv - m_new[r]);
        rs[r] += p;
        Ps[wid][r + halfw * 8][j * 16 + ln] = (_Float16)p;
      }
#pragma unroll
    for (int r = 0; r < 8; ++r) {
      l_st[r] = l_st[r] * alpha[r] + half_sum(rs[r]);
      m_st[r] = m_new[r];
    }
    // rescale running O
#pragma unroll
    for (int f = 0; f < 8; ++f)
#pragma unroll
      for (int r = 0; r < 8; ++r) o[f][r] *= alpha[r];

    // same-wave DS RAW: drain LDS stores before reading P fragments
    asm volatile("s_wait_dscnt 0" ::: "memory");

    // O += P @ V   (8 N-frags x 2 K-chain)
    v16h pa0 = load_a_frag(&Ps[wid][0][0], 72, 0, 0,  lane);
    v16h pa1 = load_a_frag(&Ps[wid][0][0], 72, 0, 32, lane);
#pragma unroll
    for (int f = 0; f < 8; ++f) {
      v16h bv0 = load_b_frag_kn(&Vs[0][0], 136, 0,  f * 16, lane);
      v16h bv1 = load_b_frag_kn(&Vs[0][0], 136, 32, f * 16, lane);
      o[f] = wmma_f16(pa0, bv0, o[f]);
      o[f] = wmma_f16(pa1, bv1, o[f]);
    }
  }

  // normalize + write merged (B,S,D) f32
#pragma unroll
  for (int r = 0; r < 8; ++r) l_st[r] = 1.0f / l_st[r];
  float* ob = attn + ((size_t)b * S) * D + (size_t)h * DH;
#pragma unroll
  for (int f = 0; f < 8; ++f)
#pragma unroll
    for (int r = 0; r < 8; ++r) {
      int srow = q0 + wid * 16 + r + halfw * 8;
      int dcol = f * 16 + ln;
      ob[(size_t)srow * D + dcol] = o[f][r] * l_st[r];
    }
}

// ---------------------------------------------------------------------------
// Kernel 3: residual + LayerNorm (one 256-thread block per row).
// ---------------------------------------------------------------------------
__global__ __launch_bounds__(256) void mha_ln(
    const float* __restrict__ attn, const float* __restrict__ qin,
    const float* __restrict__ gamma, const float* __restrict__ beta,
    float* __restrict__ out) {
  const int row = blockIdx.x;
  const float* a = attn + (size_t)row * D;
  const float* q = qin + (size_t)row * D;
  float* o = out + (size_t)row * D;
  const int tid = threadIdx.x, wid = tid >> 5, lane = tid & 31;

  float x[8];
  float s = 0.f, s2 = 0.f;
#pragma unroll
  for (int i = 0; i < 8; ++i) {
    int c = tid + i * 256;
    float v = a[c] + q[c];
    x[i] = v; s += v; s2 += v * v;
  }
#pragma unroll
  for (int m = 1; m < 32; m <<= 1) {
    s += __shfl_xor(s, m, 32);
    s2 += __shfl_xor(s2, m, 32);
  }
  __shared__ float red[2][8];
  if (lane == 0) { red[0][wid] = s; red[1][wid] = s2; }
  __syncthreads();
  if (wid == 0) {
    s  = (lane < 8) ? red[0][lane] : 0.f;
    s2 = (lane < 8) ? red[1][lane] : 0.f;
#pragma unroll
    for (int m = 1; m < 8; m <<= 1) {
      s += __shfl_xor(s, m, 32);
      s2 += __shfl_xor(s2, m, 32);
    }
    if (lane == 0) { red[0][0] = s; red[1][0] = s2; }
  }
  __syncthreads();
  const float mean = red[0][0] * (1.0f / D);
  const float var  = red[1][0] * (1.0f / D) - mean * mean;
  const float rinv = rsqrtf(var + 1e-5f);
#pragma unroll
  for (int i = 0; i < 8; ++i) {
    int c = tid + i * 256;
    o[c] = (x[i] - mean) * rinv * gamma[c] + beta[c];
  }
}

// ---------------------------------------------------------------------------
extern "C" void kernel_launch(void* const* d_in, const int* in_sizes, int n_in,
                              void* d_out, int out_size, void* d_ws, size_t ws_size,
                              hipStream_t stream) {
  const float* qin  = (const float*)d_in[0];
  const float* kvin = (const float*)d_in[1];
  const unsigned char* mask = (const unsigned char*)d_in[2];
  const float* Wq = (const float*)d_in[3];
  const float* Wk = (const float*)d_in[4];
  const float* Wv = (const float*)d_in[5];
  const float* gamma = (const float*)d_in[6];
  const float* beta  = (const float*)d_in[7];
  float* out = (float*)d_out;

  const size_t nQKV = (size_t)Bb * H * S * DH;   // 8,388,608 elems
  _Float16* Qh = (_Float16*)d_ws;
  _Float16* Kh = Qh + nQKV;
  _Float16* Vh = Kh + nQKV;
  float* attn = (float*)(Vh + nQKV);

  dim3 gp(D / 128, (Bb * S) / 64, 3);            // (16, 64, 3)
  mha_proj<<<gp, 256, 0, stream>>>(qin, kvin, Wq, Wk, Wv, Qh, Kh, Vh);

  dim3 ga(S / 64, H, Bb);                        // (32, 16, 2)
  mha_attn<<<ga, 128, 0, stream>>>(Qh, Kh, Vh, mask, attn);

  mha_ln<<<Bb * S, 256, 0, stream>>>(attn, qin, gamma, beta, out);
}
